// TransformerLayer_44959717655014
// MI455X (gfx1250) — matrix operand order repaired and checked by reading, not compiled, not run on hardware
//
#include <hip/hip_runtime.h>
#include <hip/hip_bf16.h>
#include <math.h>

// ---------------------------------------------------------------- constants
#define Bc   8
#define Sq   1024
#define Dm   1024
#define Hh   16
#define HDm  64
#define BHn  128           // B*H
#define ROWS (Bc * Sq)     // 8192

typedef __attribute__((ext_vector_type(16))) _Float16 v16h;
typedef __attribute__((ext_vector_type(8)))  _Float16 v8h;
typedef __attribute__((ext_vector_type(8)))  float    v8f;
typedef __attribute__((ext_vector_type(4)))  unsigned int v4u;
typedef __attribute__((ext_vector_type(8)))  int      v8i;
typedef __attribute__((ext_vector_type(4)))  int      v4i;

#define WMMA_F32_F16(a, b, c) \
  __builtin_amdgcn_wmma_f32_16x16x32_f16(false, (a), false, (b), (short)0, (c), false, false)

#if __has_builtin(__builtin_amdgcn_tensor_load_to_lds) && \
    __has_builtin(__builtin_amdgcn_s_wait_tensorcnt)
#define HAVE_TDM 1
#else
#define HAVE_TDM 0
#endif

// ------------------------------------------------------- fragment builders
// A-matrix 16x32 f16 (MxK): lane l -> m=l&15, h=l>>4.
// elements 0..7  = K 8h..8h+7 ; elements 8..15 = K 16+8h..16+8h+7
__device__ __forceinline__ v16h frag_a_h(const _Float16* base, int ld) {
  int lane = threadIdx.x & 31;
  int m = lane & 15, h = lane >> 4;
  const _Float16* p = base + (size_t)m * ld + 8 * h;
  v16h a;
  *((v8h*)&a)       = *((const v8h*)p);
  *(((v8h*)&a) + 1) = *((const v8h*)(p + 16));
  return a;
}

// Same A layout, f32 source converted on the fly.
__device__ __forceinline__ v16h frag_a_f32(const float* base, int ld) {
  int lane = threadIdx.x & 31;
  int m = lane & 15, h = lane >> 4;
  const float* p = base + (size_t)m * ld + 8 * h;
  v16h a;
#pragma unroll
  for (int i = 0; i < 8; ++i) a[i] = (_Float16)p[i];
#pragma unroll
  for (int i = 0; i < 8; ++i) a[8 + i] = (_Float16)p[16 + i];
  return a;
}

// B-matrix 32x16 f16 (KxN), element (k,n) at base[n*ld + k] (N-major).
// lane l -> n=l&15, kh=l>>4; elements 0..7 = K 8kh..8kh+7 ; 8..15 = K 16+8kh..16+8kh+7
__device__ __forceinline__ v16h frag_b_h(const _Float16* base, int ld) {
  int lane = threadIdx.x & 31;
  int n = lane & 15, kh = lane >> 4;
  const _Float16* p = base + (size_t)n * ld + 8 * kh;
  v16h b;
  *((v8h*)&b)       = *((const v8h*)p);
  *(((v8h*)&b) + 1) = *((const v8h*)(p + 16));
  return b;
}

__device__ __forceinline__ float red16_sum(float v) {
  v += __shfl_xor(v, 1);  v += __shfl_xor(v, 2);
  v += __shfl_xor(v, 4);  v += __shfl_xor(v, 8);
  return v;
}
__device__ __forceinline__ float red16_max(float v) {
  v = fmaxf(v, __shfl_xor(v, 1)); v = fmaxf(v, __shfl_xor(v, 2));
  v = fmaxf(v, __shfl_xor(v, 4)); v = fmaxf(v, __shfl_xor(v, 8));
  return v;
}

// ---------------------------------------------------------------- TDM
#if HAVE_TDM
// Build a 2D D# (groups 0/1 per CDNA5 ISA) and issue
// tensor_load_to_lds. f16 elements (data_size code 1). Assumes the kernel's
// single __shared__ allocation starts at LDS offset 0.
__device__ __forceinline__ void tdm_load_2d(unsigned int lds_byte,
                                            const void* gaddr,
                                            unsigned int tile_d0,
                                            unsigned int tile_d1,
                                            unsigned int stride0_elems,
                                            unsigned int tensor_d0,
                                            unsigned int tensor_d1) {
  unsigned long long ga = (unsigned long long)(uintptr_t)gaddr;
  v4u g0;
  g0[0] = 1u;                                      // count=1 user descriptor
  g0[1] = lds_byte;                                // lds_addr
  g0[2] = (unsigned int)(ga & 0xFFFFFFFFu);        // global_addr[31:0]
  g0[3] = (unsigned int)((ga >> 32) & 0x01FFFFFFu) // global_addr[56:32]
          | (2u << 30);                            // type = 2 (image)
  v8i g1;
  g1[0] = (int)(1u << 16);                                       // data_size=2B
  g1[1] = (int)((tensor_d0 & 0xFFFFu) << 16);                    // td0[15:0]
  g1[2] = (int)((tensor_d0 >> 16) | ((tensor_d1 & 0xFFFFu) << 16));
  g1[3] = (int)((tensor_d1 >> 16) | (tile_d0 << 16));            // tile_dim0
  g1[4] = (int)tile_d1;                                          // tile_dim1
  g1[5] = (int)stride0_elems;                                    // stride0 lo
  g1[6] = 0;
  g1[7] = 0;
  v4i z4 = {0, 0, 0, 0};
#if __clang_major__ >= 23
  v8i z8 = {0, 0, 0, 0, 0, 0, 0, 0};
  __builtin_amdgcn_tensor_load_to_lds(g0, g1, z4, z4, z8, 0);
#else
  __builtin_amdgcn_tensor_load_to_lds(g0, g1, z4, z4, 0);
#endif
}
#endif

// Fallback cooperative staging (also documents TDM semantics)
__device__ __forceinline__ void stage_copy_2d(_Float16* dst, const _Float16* src,
                                              int rows, int cols, int ld) {
  for (int i = threadIdx.x; i < rows * cols / 8; i += blockDim.x) {
    int r = (i * 8) / cols, c = (i * 8) % cols;
    *(v8h*)(dst + r * cols + c) = *(const v8h*)(src + (size_t)r * ld + c);
  }
}

// ---------------------------------------------------------------- kernels
__global__ void cvt_f32_h_kernel(const float* __restrict__ src,
                                 _Float16* __restrict__ dst, int n) {
  int i = blockIdx.x * blockDim.x + threadIdx.x;
  if (i < n) dst[i] = (_Float16)src[i];
}

// one 16x64 head-projection tile: y = x @ W^T + b (W is 64x64 (out,in))
template <int MODE>  // 0: L2-normalize rows, row-major f16 ; 1: transposed d-major
__device__ __forceinline__ void proj_tile(const float* xbase, const _Float16* W,
                                          const float* bias, _Float16* dst, int s0) {
  int lane = threadIdx.x & 31;
  int n16 = lane & 15, half = lane >> 4;
  v8f acc[4];
#pragma unroll
  for (int t = 0; t < 4; ++t)
#pragma unroll
    for (int r = 0; r < 8; ++r) acc[t][r] = 0.0f;
#pragma unroll
  for (int ks = 0; ks < 2; ++ks) {
    v16h a = frag_a_f32(xbase + 32 * ks, Dm);
#pragma unroll
    for (int t = 0; t < 4; ++t) {
      v16h b = frag_b_h(W + t * 16 * HDm + 32 * ks, HDm);
      acc[t] = WMMA_F32_F16(a, b, acc[t]);
    }
  }
  float val[4][8];
#pragma unroll
  for (int t = 0; t < 4; ++t)
#pragma unroll
    for (int r = 0; r < 8; ++r) val[t][r] = acc[t][r] + bias[t * 16 + n16];

  if (MODE == 0) {
    float scale[8];
#pragma unroll
    for (int r = 0; r < 8; ++r) {
      float ss = 0.0f;
#pragma unroll
      for (int t = 0; t < 4; ++t) ss += val[t][r] * val[t][r];
      ss = red16_sum(ss);
      scale[r] = 1.0f / fmaxf(sqrtf(ss), 1e-12f);
    }
#pragma unroll
    for (int t = 0; t < 4; ++t)
#pragma unroll
      for (int r = 0; r < 8; ++r)
        dst[(size_t)(r + 8 * half) * HDm + t * 16 + n16] =
            (_Float16)(val[t][r] * scale[r]);
  } else {
#pragma unroll
    for (int t = 0; t < 4; ++t)
#pragma unroll
      for (int r = 0; r < 8; ++r)
        dst[(size_t)(t * 16 + n16) * Sq + (s0 + r + 8 * half)] = (_Float16)val[t][r];
  }
}

__global__ __launch_bounds__(256) void proj_kernel(
    const float* __restrict__ q, const float* __restrict__ k, const float* __restrict__ v,
    const _Float16* __restrict__ Wqh, const _Float16* __restrict__ Wkh,
    const _Float16* __restrict__ Wvh,
    const float* __restrict__ bq, const float* __restrict__ bk, const float* __restrict__ bv,
    _Float16* __restrict__ qn, _Float16* __restrict__ kn, _Float16* __restrict__ vT) {
  int wave = threadIdx.x >> 5;
  int blk = blockIdx.x;
  int bh = blk >> 3;
  int s0 = (blk & 7) * 128 + wave * 16;
  int b = bh >> 4, h = bh & 15;
  size_t xoff = ((size_t)(b * Sq + s0)) * Dm + h * HDm;
  proj_tile<0>(q + xoff, Wqh, bq, qn + ((size_t)bh * Sq + s0) * HDm, s0);
  proj_tile<0>(k + xoff, Wkh, bk, kn + ((size_t)bh * Sq + s0) * HDm, s0);
  proj_tile<1>(v + xoff, Wvh, bv, vT + (size_t)bh * HDm * Sq, s0);
}

// flash attention, TDM-staged K/V chunks (shared by all 8 waves), double buffered.
// LDS map (elems): K[2][2048] @0, V[2][2048] @4096, P[8][512] @8192
__global__ __launch_bounds__(256) void attn_kernel(
    const _Float16* __restrict__ qn, const _Float16* __restrict__ kn,
    const _Float16* __restrict__ vT, float* __restrict__ heads) {
  __shared__ __align__(16) _Float16 smA[12288];
  _Float16* Ksm = smA;
  _Float16* Vsm = smA + 4096;
  int wave = threadIdx.x >> 5, lane = threadIdx.x & 31;
  int n16 = lane & 15, half = lane >> 4;
  int blk = blockIdx.x;
  int bh = blk >> 3;
  int s0 = (blk & 7) * 128 + wave * 16;
  const _Float16* qbase = qn + ((size_t)bh * Sq + s0) * HDm;
  const _Float16* kbase = kn + (size_t)bh * Sq * HDm;
  const _Float16* vbase = vT + (size_t)bh * HDm * Sq;
  _Float16* P = smA + 8192 + wave * 512;

  v16h aq0 = frag_a_h(qbase, HDm);
  v16h aq1 = frag_a_h(qbase + 32, HDm);

  v8f o[4];
  float mrow[8], lrow[8];
#pragma unroll
  for (int t = 0; t < 4; ++t)
#pragma unroll
    for (int r = 0; r < 8; ++r) o[t][r] = 0.0f;
#pragma unroll
  for (int r = 0; r < 8; ++r) { mrow[r] = -3.0e38f; lrow[r] = 0.0f; }

#if HAVE_TDM
  if (wave == 0) {
    tdm_load_2d(0,    kbase, 2048, 1, 2048, 2048, 1);      // K chunk: contiguous
    tdm_load_2d(8192, vbase, 32, 64, Sq, Sq, HDm);         // V chunk: strided
    __builtin_amdgcn_s_wait_tensorcnt(0);
  }
  __syncthreads();
#else
  stage_copy_2d(Ksm, kbase, 1, 2048, 2048);
  stage_copy_2d(Vsm, vbase, 64, 32, Sq);
  __syncthreads();
#endif

  for (int c = 0; c < Sq / 32; ++c) {
    int buf = c & 1;
    int kt = c * 32;
#if HAVE_TDM
    if (c + 1 < Sq / 32 && wave == 0) {
      tdm_load_2d((unsigned)(buf ^ 1) * 4096,
                  kbase + (size_t)(kt + 32) * HDm, 2048, 1, 2048, 2048, 1);
      tdm_load_2d(8192u + (unsigned)(buf ^ 1) * 4096,
                  vbase + (kt + 32), 32, 64, Sq, Sq, HDm);
    }
#endif
    const _Float16* Kc = Ksm + buf * 2048;
    const _Float16* Vc = Vsm + buf * 2048;

    v8f s[2];
#pragma unroll
    for (int sub = 0; sub < 2; ++sub) {
      v8f sc;
#pragma unroll
      for (int r = 0; r < 8; ++r) sc[r] = 0.0f;
      const _Float16* kb = Kc + (size_t)(16 * sub) * HDm;
      sc = WMMA_F32_F16(aq0, frag_b_h(kb, HDm), sc);
      sc = WMMA_F32_F16(aq1, frag_b_h(kb + 32, HDm), sc);
      s[sub] = sc;
    }
    float alpha[8];
#pragma unroll
    for (int r = 0; r < 8; ++r) {
      float mx = red16_max(fmaxf(s[0][r], s[1][r]));
      float mnew = fmaxf(mrow[r], mx);
      alpha[r] = __expf(mrow[r] - mnew);
      mrow[r] = mnew;
    }
    float psum[8];
#pragma unroll
    for (int r = 0; r < 8; ++r) psum[r] = 0.0f;
#pragma unroll
    for (int sub = 0; sub < 2; ++sub)
#pragma unroll
      for (int r = 0; r < 8; ++r) {
        float p = __expf(s[sub][r] - mrow[r]);
        s[sub][r] = p;
        psum[r] += p;
      }
#pragma unroll
    for (int r = 0; r < 8; ++r) lrow[r] = lrow[r] * alpha[r] + red16_sum(psum[r]);
#pragma unroll
    for (int sub = 0; sub < 2; ++sub)
#pragma unroll
      for (int r = 0; r < 8; ++r)
        P[(r + 8 * half) * 32 + sub * 16 + n16] = (_Float16)s[sub][r];
#pragma unroll
    for (int t = 0; t < 4; ++t)
#pragma unroll
      for (int r = 0; r < 8; ++r) o[t][r] *= alpha[r];
    v16h ap = frag_a_h(P, 32);
#pragma unroll
    for (int t = 0; t < 4; ++t) {
      v16h bv = frag_b_h(Vc + (size_t)(t * 16) * 32, 32);
      o[t] = WMMA_F32_F16(ap, bv, o[t]);
    }
#if HAVE_TDM
    if (wave == 0) __builtin_amdgcn_s_wait_tensorcnt(0);
    __syncthreads();
#else
    __syncthreads();
    if (c + 1 < Sq / 32) {
      stage_copy_2d(Ksm + (buf ^ 1) * 2048, kbase + (size_t)(kt + 32) * HDm, 1, 2048, 2048);
      stage_copy_2d(Vsm + (buf ^ 1) * 2048, vbase + (kt + 32), 64, 32, Sq);
      __syncthreads();
    }
#endif
  }
  int b = bh >> 4, h = bh & 15;
  float* obase = heads + ((size_t)(b * Sq + s0)) * Dm + h * HDm;
#pragma unroll
  for (int t = 0; t < 4; ++t)
#pragma unroll
    for (int r = 0; r < 8; ++r)
      obase[(size_t)(r + 8 * half) * Dm + t * 16 + n16] = o[t][r] / lrow[r];
}

// y = LayerNorm(resid + x); write f32 and f16 copies
__global__ __launch_bounds__(256) void addnorm_kernel(
    const float* __restrict__ resid, const float* __restrict__ x,
    const float* __restrict__ g, const float* __restrict__ bt,
    float* __restrict__ yf, _Float16* __restrict__ yh) {
  __shared__ float red[2][8];
  int row = blockIdx.x;
  const float* xr = x + (size_t)row * Dm;
  const float* rr = resid + (size_t)row * Dm;
  float vals[4], s1 = 0.0f, s2 = 0.0f;
#pragma unroll
  for (int i = 0; i < 4; ++i) {
    int c = threadIdx.x + i * 256;
    float t = rr[c] + xr[c];
    vals[i] = t; s1 += t; s2 += t * t;
  }
#pragma unroll
  for (int m = 1; m < 32; m <<= 1) { s1 += __shfl_xor(s1, m); s2 += __shfl_xor(s2, m); }
  int wave = threadIdx.x >> 5, lane = threadIdx.x & 31;
  if (lane == 0) { red[0][wave] = s1; red[1][wave] = s2; }
  __syncthreads();
  float t1 = 0.0f, t2 = 0.0f;
#pragma unroll
  for (int w = 0; w < 8; ++w) { t1 += red[0][w]; t2 += red[1][w]; }
  float mu = t1 * (1.0f / Dm);
  float var = t2 * (1.0f / Dm) - mu * mu;
  float rs = rsqrtf(var + 1e-5f);
#pragma unroll
  for (int i = 0; i < 4; ++i) {
    int c = threadIdx.x + i * 256;
    float yn = (vals[i] - mu) * rs * g[c] + bt[c];
    yf[(size_t)row * Dm + c] = yn;
    yh[(size_t)row * Dm + c] = (_Float16)yn;
  }
}

// Shared FFN mainloop: M=32 rows/block (2 M-tiles), N=128/wave (8 N-tiles),
// A tile TDM-staged into LDS (double-buffered 64-wide K chunks), B from L2.
// LDS map (bytes): A[2][4096] @0, red1 @8192, red2 @9216
__device__ __forceinline__ void ffn_mainloop(const _Float16* abase,
                                             const _Float16* W, int n0,
                                             _Float16* Asm, v8f acc[2][8]) {
#pragma unroll
  for (int mt = 0; mt < 2; ++mt)
#pragma unroll
    for (int t = 0; t < 8; ++t)
#pragma unroll
      for (int r = 0; r < 8; ++r) acc[mt][t][r] = 0.0f;
  int wave = threadIdx.x >> 5;
#if HAVE_TDM
  if (wave == 0) {
    tdm_load_2d(0, abase, 64, 32, Dm, Dm, 32);
    __builtin_amdgcn_s_wait_tensorcnt(0);
  }
  __syncthreads();
#else
  stage_copy_2d(Asm, abase, 32, 64, Dm);
  __syncthreads();
#endif
  for (int c = 0; c < Dm / 64; ++c) {
    int buf = c & 1;
#if HAVE_TDM
    if (c + 1 < Dm / 64 && wave == 0)
      tdm_load_2d((unsigned)(buf ^ 1) * 4096, abase + (c + 1) * 64, 64, 32, Dm, Dm, 32);
#endif
    __builtin_prefetch(W + (size_t)n0 * Dm + (c + 1) * 64, 0, 1);
#pragma unroll
    for (int ks = 0; ks < 2; ++ks) {
      int k0 = c * 64 + ks * 32;
      v16h a0 = frag_a_h(Asm + buf * 2048 + ks * 32, 64);
      v16h a1 = frag_a_h(Asm + buf * 2048 + 16 * 64 + ks * 32, 64);
#pragma unroll
      for (int t = 0; t < 8; ++t) {
        v16h b = frag_b_h(W + (size_t)(n0 + t * 16) * Dm + k0, Dm);
        acc[0][t] = WMMA_F32_F16(a0, b, acc[0][t]);
        acc[1][t] = WMMA_F32_F16(a1, b, acc[1][t]);
      }
    }
#if HAVE_TDM
    if (wave == 0) __builtin_amdgcn_s_wait_tensorcnt(0);
    __syncthreads();
#else
    __syncthreads();
    if (c + 1 < Dm / 64) {
      stage_copy_2d(Asm + (buf ^ 1) * 2048, abase + (c + 1) * 64, 32, 64, Dm);
      __syncthreads();
    }
#endif
  }
}

// FFN GEMM1: H = gelu(A @ W1^T + b1), f16 out
__global__ __launch_bounds__(256) void ffn1_kernel(
    const _Float16* __restrict__ A, const _Float16* __restrict__ W,
    const float* __restrict__ bias, _Float16* __restrict__ Hout) {
  __shared__ __align__(16) char smem[8192];
  int wave = threadIdx.x >> 5, lane = threadIdx.x & 31;
  int n16 = lane & 15, half = lane >> 4;
  int row0 = blockIdx.x * 32;
  int n0 = wave * 128;
  v8f acc[2][8];
  ffn_mainloop(A + (size_t)row0 * Dm, W, n0, (_Float16*)smem, acc);
#pragma unroll
  for (int mt = 0; mt < 2; ++mt)
#pragma unroll
    for (int t = 0; t < 8; ++t)
#pragma unroll
      for (int r = 0; r < 8; ++r) {
        int n = n0 + t * 16 + n16;
        float xg = acc[mt][t][r] + bias[n];
        float ge = 0.5f * xg * (1.0f + erff(xg * 0.70710678118654752f));
        Hout[(size_t)(row0 + mt * 16 + r + 8 * half) * Dm + n] = (_Float16)ge;
      }
}

// FFN GEMM2 + residual + LayerNorm2, f32 out
__global__ __launch_bounds__(256) void ffn2_kernel(
    const _Float16* __restrict__ A, const _Float16* __restrict__ W,
    const float* __restrict__ bias, const float* __restrict__ resid,
    const float* __restrict__ g, const float* __restrict__ bt,
    float* __restrict__ out) {
  __shared__ __align__(16) char smem[8192 + 1024 + 1024];
  float* red1 = (float*)(smem + 8192);   // [32 rows][8 waves]
  float* red2 = (float*)(smem + 9216);
  int wave = threadIdx.x >> 5, lane = threadIdx.x & 31;
  int n16 = lane & 15, half = lane >> 4;
  int row0 = blockIdx.x * 32;
  int n0 = wave * 128;
  v8f acc[2][8];
  ffn_mainloop(A + (size_t)row0 * Dm, W, n0, (_Float16*)smem, acc);
  // x = acc + bias + residual (in place)
#pragma unroll
  for (int mt = 0; mt < 2; ++mt)
#pragma unroll
    for (int t = 0; t < 8; ++t)
#pragma unroll
      for (int r = 0; r < 8; ++r) {
        int n = n0 + t * 16 + n16;
        size_t row = row0 + mt * 16 + r + 8 * half;
        acc[mt][t][r] = acc[mt][t][r] + bias[n] + resid[row * Dm + n];
      }
#pragma unroll
  for (int mt = 0; mt < 2; ++mt)
#pragma unroll
    for (int r = 0; r < 8; ++r) {
      float p1 = 0.0f, p2 = 0.0f;
#pragma unroll
      for (int t = 0; t < 8; ++t) { p1 += acc[mt][t][r]; p2 += acc[mt][t][r] * acc[mt][t][r]; }
      p1 = red16_sum(p1);
      p2 = red16_sum(p2);
      if (n16 == 0) {
        int rr = mt * 16 + r + 8 * half;
        red1[rr * 8 + wave] = p1;
        red2[rr * 8 + wave] = p2;
      }
    }
  __syncthreads();
#pragma unroll
  for (int mt = 0; mt < 2; ++mt)
#pragma unroll
    for (int r = 0; r < 8; ++r) {
      int rr = mt * 16 + r + 8 * half;
      float t1 = 0.0f, t2 = 0.0f;
#pragma unroll
      for (int w = 0; w < 8; ++w) { t1 += red1[rr * 8 + w]; t2 += red2[rr * 8 + w]; }
      float mu = t1 * (1.0f / Dm);
      float var = t2 * (1.0f / Dm) - mu * mu;
      float rs = rsqrtf(var + 1e-5f);
#pragma unroll
      for (int t = 0; t < 8; ++t) {
        int n = n0 + t * 16 + n16;
        out[(size_t)(row0 + rr) * Dm + n] = (acc[mt][t][r] - mu) * rs * g[n] + bt[n];
      }
    }
}

// ---------------------------------------------------------------- launcher
extern "C" void kernel_launch(void* const* d_in, const int* in_sizes, int n_in,
                              void* d_out, int out_size, void* d_ws, size_t ws_size,
                              hipStream_t stream) {
  (void)in_sizes; (void)n_in; (void)out_size; (void)ws_size;
  const float* q   = (const float*)d_in[0];
  const float* k   = (const float*)d_in[1];
  const float* v   = (const float*)d_in[2];
  const float* Wq  = (const float*)d_in[3];  const float* bq = (const float*)d_in[4];
  const float* Wk  = (const float*)d_in[5];  const float* bk = (const float*)d_in[6];
  const float* Wv  = (const float*)d_in[7];  const float* bv = (const float*)d_in[8];
  const float* W1  = (const float*)d_in[9];  const float* b1 = (const float*)d_in[10];
  const float* W2  = (const float*)d_in[11]; const float* b2 = (const float*)d_in[12];
  const float* ln1g = (const float*)d_in[13]; const float* ln1b = (const float*)d_in[14];
  const float* ln2g = (const float*)d_in[15]; const float* ln2b = (const float*)d_in[16];
  float* out = (float*)d_out;

  char* ws = (char*)d_ws;
  size_t off = 0;
  auto carve = [&](size_t bytes) -> void* {
    void* p = ws + off;
    off = (off + bytes + 255) & ~(size_t)255;
    return p;
  };
  _Float16* Wqh = (_Float16*)carve(HDm * HDm * 2);
  _Float16* Wkh = (_Float16*)carve(HDm * HDm * 2);
  _Float16* Wvh = (_Float16*)carve(HDm * HDm * 2);
  _Float16* W1h = (_Float16*)carve((size_t)Dm * Dm * 2);
  _Float16* W2h = (_Float16*)carve((size_t)Dm * Dm * 2);
  _Float16* qnh = (_Float16*)carve((size_t)BHn * Sq * HDm * 2);
  _Float16* knh = (_Float16*)carve((size_t)BHn * Sq * HDm * 2);
  _Float16* vTh = (_Float16*)carve((size_t)BHn * Sq * HDm * 2);
  float*    y1f = (float*)carve((size_t)ROWS * Dm * 4);
  _Float16* y1h = (_Float16*)carve((size_t)ROWS * Dm * 2);
  _Float16* hbf = (_Float16*)carve((size_t)ROWS * Dm * 2);

  cvt_f32_h_kernel<<<(HDm * HDm + 255) / 256, 256, 0, stream>>>(Wq, Wqh, HDm * HDm);
  cvt_f32_h_kernel<<<(HDm * HDm + 255) / 256, 256, 0, stream>>>(Wk, Wkh, HDm * HDm);
  cvt_f32_h_kernel<<<(HDm * HDm + 255) / 256, 256, 0, stream>>>(Wv, Wvh, HDm * HDm);
  cvt_f32_h_kernel<<<(Dm * Dm + 255) / 256, 256, 0, stream>>>(W1, W1h, Dm * Dm);
  cvt_f32_h_kernel<<<(Dm * Dm + 255) / 256, 256, 0, stream>>>(W2, W2h, Dm * Dm);

  proj_kernel<<<BHn * (Sq / 128), 256, 0, stream>>>(q, k, v, Wqh, Wkh, Wvh,
                                                    bq, bk, bv, qnh, knh, vTh);
  attn_kernel<<<BHn * (Sq / 128), 256, 0, stream>>>(qnh, knh, vTh, out);
  addnorm_kernel<<<ROWS, 256, 0, stream>>>(v, out, ln1g, ln1b, y1f, y1h);
  ffn1_kernel<<<ROWS / 32, 256, 0, stream>>>(y1h, W1h, b1, hbf);
  ffn2_kernel<<<ROWS / 32, 256, 0, stream>>>(hbf, W2h, b2, y1f, ln2g, ln2b, out);
}
